// TokenQKAttention_77111842832414
// MI455X (gfx1250) — matrix operand order repaired and checked by reading, not compiled, not run on hardware
//
#include <hip/hip_runtime.h>

// ---------------------------------------------------------------------------
// Problem constants (reference: T=4, B=32, C=512, H=W=16 -> N=256, heads=8)
// ---------------------------------------------------------------------------
constexpr int T_  = 4;
constexpr int B_  = 32;
constexpr int C_  = 512;
constexpr int N_  = 256;
constexpr int H_  = 8;
constexpr int HD_ = 64;
constexpr int TB_ = T_ * B_;          // 128 batched GEMMs
constexpr int QKC = 2 * C_;           // 1024: concat(q,k) output channels

// GEMM tiling: block tile 128(M) x 128(N), K-step 32 (== WMMA K), 8 waves,
// each wave owns a 32(M) x 64(N) sub-tile -> 8 WMMAs per K-step per wave.
constexpr int BM  = 128;
constexpr int BNT = 128;
constexpr int BK  = 32;
constexpr int LDA = 40;               // padded LDS K-stride (80B, 16B-aligned rows)

typedef __attribute__((ext_vector_type(16))) __bf16 v16bf;
typedef __attribute__((ext_vector_type(8)))  float  v8f;

// ---------------------------------------------------------------------------
// WMMA helper: D = A(16x32 bf16) * B(32x16 bf16) + C(16x16 f32)
// ---------------------------------------------------------------------------
__device__ __forceinline__ v8f wmma_bf16(v16bf a, v16bf b, v8f c) {
    return __builtin_amdgcn_wmma_f32_16x16x32_bf16(
        /*neg_a=*/false, a, /*neg_b=*/false, b,
        /*c_mod=*/(short)0, c, /*reuse_a=*/false, /*reuse_b=*/false);
}

// A-fragment (16x32, M=lane&15): VGPR0-7 hold K = 16*half + 0..15
// (contiguous order, matching ld_fragB)
__device__ __forceinline__ v16bf ld_fragA(const __bf16* row, int half) {
    union { v16bf v; uint4 q[2]; } u;
    u.q[0] = *(const uint4*)(row + half * 16);
    u.q[1] = *(const uint4*)(row + half * 16 + 8);
    return u.v;
}

// B-fragment (32x16, N=lane&15): lanes 0-15 hold K=0..15, lanes 16-31 K=16..31,
// 2 packed K per VGPR. sB stores B transposed: row = n, contiguous in K.
__device__ __forceinline__ v16bf ld_fragB(const __bf16* row, int half) {
    union { v16bf v; uint4 q[2]; } u;
    u.q[0] = *(const uint4*)(row + half * 16);
    u.q[1] = *(const uint4*)(row + half * 16 + 8);
    return u.v;
}

// ---------------------------------------------------------------------------
// Tile loaders: global -> LDS with fp32->bf16 conversion, B transposed (K-major)
// ---------------------------------------------------------------------------
__device__ __forceinline__ void load_tileA(const float* Atile, __bf16* sA, int tid) {
    // BM x BK fp32 = 1024 float4, 4 per thread
#pragma unroll
    for (int i = 0; i < 4; ++i) {
        int idx = tid + i * 256;
        int row = idx >> 3;        // 8 float4 per 32-float row
        int c4  = idx & 7;
        float4 f = *(const float4*)(Atile + (size_t)row * C_ + c4 * 4);
        __bf16* d = sA + row * LDA + c4 * 4;
        d[0] = (__bf16)f.x; d[1] = (__bf16)f.y;
        d[2] = (__bf16)f.z; d[3] = (__bf16)f.w;
    }
}

__device__ __forceinline__ void load_tileB(const float* Btile, __bf16* sB, int tid) {
    // BK x BNT fp32 (32 x 128). Each thread handles K rows {2kp, 2kp+1} for 8 n,
    // packing bf16 pairs (k, k+1) -> single 4-byte LDS store (matches frag layout).
    const int kp = tid >> 4;              // 0..15  -> k0 = 2kp
    const int n0 = (tid & 15) * 8;        // 0..120
    const float* r0 = Btile + (size_t)(2 * kp) * N_ + n0;
    const float* r1 = r0 + N_;
    float4 a0 = *(const float4*)(r0);
    float4 a1 = *(const float4*)(r0 + 4);
    float4 b0 = *(const float4*)(r1);
    float4 b1 = *(const float4*)(r1 + 4);
    const float f0[8] = {a0.x, a0.y, a0.z, a0.w, a1.x, a1.y, a1.z, a1.w};
    const float f1[8] = {b0.x, b0.y, b0.z, b0.w, b1.x, b1.y, b1.z, b1.w};
#pragma unroll
    for (int i = 0; i < 8; ++i) {
        union { unsigned u; __bf16 h[2]; } p;
        p.h[0] = (__bf16)f0[i];
        p.h[1] = (__bf16)f1[i];
        *(unsigned*)(sB + (n0 + i) * LDA + 2 * kp) = p.u;
    }
}

__device__ __forceinline__ void load_tileB(const __bf16* Btile, __bf16* sB, int tid) {
    // bf16 input (y matrix): same pairing, no conversion needed
    const int kp = tid >> 4;
    const int n0 = (tid & 15) * 8;
    union { uint4 q; __bf16 h[8]; } r0, r1;
    r0.q = *(const uint4*)(Btile + (size_t)(2 * kp) * N_ + n0);
    r1.q = *(const uint4*)(Btile + (size_t)(2 * kp + 1) * N_ + n0);
#pragma unroll
    for (int i = 0; i < 8; ++i) {
        union { unsigned u; __bf16 h[2]; } p;
        p.h[0] = r0.h[i];
        p.h[1] = r1.h[i];
        *(unsigned*)(sB + (n0 + i) * LDA + 2 * kp) = p.u;
    }
}

// ---------------------------------------------------------------------------
// Batched GEMM: out[batch, o, n] = sum_c A[o,c] * Bsrc[batch, c, n]  (+bias)
// MODE 0: A = concat(q_w rows 0-511, k_w rows 512-1023), B = x fp32, out bf16
// MODE 1: A = proj_w (512 rows), B = y bf16, out f32 + bias
// grid: (N_/BNT, Mtiles, TB_), block: 256 (8 waves, 4x2 wave grid of 32x64)
// ---------------------------------------------------------------------------
template <int MODE>
__global__ void __launch_bounds__(256)
gemm_kernel(const float* Aq, const float* Ak,
            const float* Bf, const __bf16* Bh, size_t bstride,
            __bf16* outBf, float* outF, const float* bias) {
    __shared__ __bf16 sA[2][BM * LDA];
    __shared__ __bf16 sB[2][BNT * LDA];

    const int tid   = threadIdx.x;
    const int ntile = blockIdx.x;
    const int mtile = blockIdx.y;
    const int batch = blockIdx.z;

    const float* A;
    int mrowbase;
    if constexpr (MODE == 0) {
        A = (mtile < 4) ? Aq : Ak;           // concat(q_w, k_w)
        mrowbase = (mtile & 3) * BM;
    } else {
        A = Aq;
        mrowbase = mtile * BM;
    }
    const float*  BbF = (MODE == 0) ? (Bf + (size_t)batch * bstride + (size_t)ntile * BNT) : nullptr;
    const __bf16* BbH = (MODE == 1) ? (Bh + (size_t)batch * bstride + (size_t)ntile * BNT) : nullptr;

    const int lane = tid & 31;
    const int wv   = tid >> 5;
    const int wm   = (wv >> 1) * 32;          // wave M offset (0,32,64,96)
    const int wn   = (wv & 1) * 64;           // wave N offset (0,64)
    const int half = lane >> 4;
    const int lr   = lane & 15;

    v8f acc[2][4] = {};

    // prologue: stage K-tile 0
    load_tileA(A + (size_t)mrowbase * C_, sA[0], tid);
    if constexpr (MODE == 0) load_tileB(BbF, sB[0], tid);
    else                     load_tileB(BbH, sB[0], tid);
    __syncthreads();

    for (int kb = 0; kb < C_; kb += BK) {
        const int cur = (kb / BK) & 1;
        if (kb + BK < C_) {
            load_tileA(A + (size_t)mrowbase * C_ + (kb + BK), sA[cur ^ 1], tid);
            if constexpr (MODE == 0) load_tileB(BbF + (size_t)(kb + BK) * N_, sB[cur ^ 1], tid);
            else                     load_tileB(BbH + (size_t)(kb + BK) * N_, sB[cur ^ 1], tid);
        }

        const __bf16* pa = &sA[cur][(wm + lr) * LDA];
        const __bf16* pb = &sB[cur][(wn + lr) * LDA];
        v16bf a0 = ld_fragA(pa, half);
        v16bf a1 = ld_fragA(pa + 16 * LDA, half);
        v16bf b0 = ld_fragB(pb, half);
        v16bf b1 = ld_fragB(pb + 16 * LDA, half);
        v16bf b2 = ld_fragB(pb + 32 * LDA, half);
        v16bf b3 = ld_fragB(pb + 48 * LDA, half);
        acc[0][0] = wmma_bf16(a0, b0, acc[0][0]);
        acc[0][1] = wmma_bf16(a0, b1, acc[0][1]);
        acc[0][2] = wmma_bf16(a0, b2, acc[0][2]);
        acc[0][3] = wmma_bf16(a0, b3, acc[0][3]);
        acc[1][0] = wmma_bf16(a1, b0, acc[1][0]);
        acc[1][1] = wmma_bf16(a1, b1, acc[1][1]);
        acc[1][2] = wmma_bf16(a1, b2, acc[1][2]);
        acc[1][3] = wmma_bf16(a1, b3, acc[1][3]);

        __syncthreads();
    }

    // epilogue: C/D layout — VGPR j: lanes0-15 M=j, lanes16-31 M=j+8; N = lane&15
#pragma unroll
    for (int mi = 0; mi < 2; ++mi) {
#pragma unroll
        for (int ni = 0; ni < 4; ++ni) {
            const int obase = mtile * BM + wm + mi * 16 + half * 8;
            const int n     = ntile * BNT + wn + ni * 16 + lr;
#pragma unroll
            for (int j = 0; j < 8; ++j) {
                const int o = obase + j;
                if constexpr (MODE == 0) {
                    outBf[((size_t)batch * QKC + o) * N_ + n] = (__bf16)acc[mi][ni][j];
                } else {
                    outF[((size_t)batch * C_ + o) * N_ + n] = acc[mi][ni][j] + bias[o];
                }
            }
        }
    }
}

// ---------------------------------------------------------------------------
// BN batch stats: per channel over (T,B,N). grid = Cdim, block = 256 (one n each)
// ---------------------------------------------------------------------------
template <typename TIN>
__global__ void __launch_bounds__(256)
bn_stats_kernel(const TIN* x, float* mean, float* rstd, int Cdim) {
    const int c = blockIdx.x, n = threadIdx.x;
    float s = 0.f, s2 = 0.f;
    for (int sl = 0; sl < TB_; ++sl) {
        float v = (float)x[((size_t)sl * Cdim + c) * N_ + n];
        s += v; s2 += v * v;
    }
    __shared__ float rs[256], rq[256];
    rs[n] = s; rq[n] = s2;
    __syncthreads();
    for (int off = 128; off > 0; off >>= 1) {
        if (n < off) { rs[n] += rs[n + off]; rq[n] += rq[n + off]; }
        __syncthreads();
    }
    if (n == 0) {
        const float cnt = (float)(TB_ * N_);
        float m = rs[0] / cnt;
        float var = rq[0] / cnt - m * m;
        mean[c] = m;
        rstd[c] = rsqrtf(var + 1e-5f);
    }
}

// ---------------------------------------------------------------------------
// q: BN + LIF + head-sum + attn LIF. grid = B*H, block = 256 (n)
// ---------------------------------------------------------------------------
__global__ void __launch_bounds__(256)
q_attn_kernel(const __bf16* qk, const float* mean, const float* rstd,
              const float* gamma, const float* beta, __bf16* attn) {
    const int b = blockIdx.x >> 3, h = blockIdx.x & 7, n = threadIdx.x;
    float ssum[T_] = {};
    for (int d = 0; d < HD_; ++d) {
        const int c = h * HD_ + d;
        const float m = mean[c], r = rstd[c], g = gamma[c], be = beta[c];
        float v = 0.f;
#pragma unroll
        for (int t = 0; t < T_; ++t) {
            float qv = (float)qk[(((size_t)(t * B_ + b)) * QKC + c) * N_ + n];
            float xb = g * (qv - m) * r + be;
            v += (xb - v) * 0.5f;                 // TAU = 2
            float sp = (v >= 1.0f) ? 1.f : 0.f;   // V_TH = 1
            v *= (1.f - sp);
            ssum[t] += sp;
        }
    }
    float v = 0.f;
#pragma unroll
    for (int t = 0; t < T_; ++t) {
        v += (ssum[t] - v) * 0.5f;
        float a = (v >= 1.0f) ? 1.f : 0.f;
        v *= (1.f - a);
        attn[(((size_t)(t * B_ + b)) * H_ + h) * N_ + n] = (__bf16)a;
    }
}

// ---------------------------------------------------------------------------
// k: BN + LIF, y = attn * spike, written (bf16) into the q half of qkbuf.
// grid = B*C, block = 256 (n)
// ---------------------------------------------------------------------------
__global__ void __launch_bounds__(256)
k_y_kernel(const __bf16* qk, const float* mean, const float* rstd,
           const float* kg, const float* kb, const __bf16* attn, __bf16* y) {
    const int b = blockIdx.x >> 9, c = blockIdx.x & 511, n = threadIdx.x;
    const int h = c >> 6;
    const float m = mean[C_ + c], r = rstd[C_ + c], g = kg[c], be = kb[c];
    float v = 0.f;
#pragma unroll
    for (int t = 0; t < T_; ++t) {
        float kv = (float)qk[(((size_t)(t * B_ + b)) * QKC + C_ + c) * N_ + n];
        float xb = g * (kv - m) * r + be;
        v += (xb - v) * 0.5f;
        float sp = (v >= 1.0f) ? 1.f : 0.f;
        v *= (1.f - sp);
        float a = (float)attn[(((size_t)(t * B_ + b)) * H_ + h) * N_ + n];
        // y aliases q half: rows 0..511 of the [QKC, N] per-batch slab
        y[(((size_t)(t * B_ + b)) * QKC + c) * N_ + n] = (__bf16)(sp * a);
    }
}

// ---------------------------------------------------------------------------
// final: BN + LIF -> f32 spikes. grid = B*C, block = 256 (n)
// ---------------------------------------------------------------------------
__global__ void __launch_bounds__(256)
out_kernel(const float* ylin, const float* mean, const float* rstd,
           const float* g_, const float* b_, float* out) {
    const int b = blockIdx.x >> 9, c = blockIdx.x & 511, n = threadIdx.x;
    const float m = mean[c], r = rstd[c], g = g_[c], be = b_[c];
    float v = 0.f;
#pragma unroll
    for (int t = 0; t < T_; ++t) {
        const size_t idx = (((size_t)(t * B_ + b)) * C_ + c) * N_ + n;
        float xb = g * (ylin[idx] - m) * r + be;
        v += (xb - v) * 0.5f;
        float sp = (v >= 1.0f) ? 1.f : 0.f;
        v *= (1.f - sp);
        out[idx] = sp;
    }
}

// ---------------------------------------------------------------------------
// Host launcher
// ---------------------------------------------------------------------------
extern "C" void kernel_launch(void* const* d_in, const int* in_sizes, int n_in,
                              void* d_out, int out_size, void* d_ws, size_t ws_size,
                              hipStream_t stream) {
    const float* x          = (const float*)d_in[0];
    const float* q_w        = (const float*)d_in[1];
    const float* q_gamma    = (const float*)d_in[2];
    const float* q_beta     = (const float*)d_in[3];
    const float* k_w        = (const float*)d_in[4];
    const float* k_gamma    = (const float*)d_in[5];
    const float* k_beta     = (const float*)d_in[6];
    const float* proj_w     = (const float*)d_in[7];
    const float* proj_b     = (const float*)d_in[8];
    const float* proj_gamma = (const float*)d_in[9];
    const float* proj_beta  = (const float*)d_in[10];
    float* out = (float*)d_out;

    // workspace layout
    char* ws = (char*)d_ws;
    const size_t qk_bytes   = (size_t)TB_ * QKC * N_ * sizeof(__bf16);   // 64 MiB
    const size_t ylin_bytes = (size_t)TB_ * C_ * N_ * sizeof(float);     // 64 MiB
    const size_t attn_bytes = (size_t)TB_ * H_ * N_ * sizeof(__bf16);    // 0.5 MiB
    __bf16* qkbuf = (__bf16*)ws;
    float*  ylin  = (float*)(ws + qk_bytes);
    __bf16* attn  = (__bf16*)(ws + qk_bytes + ylin_bytes);
    float*  stats = (float*)(ws + qk_bytes + ylin_bytes + attn_bytes);
    float* mean_qk = stats;            // 1024
    float* rstd_qk = stats + 1024;     // 1024
    float* mean_p  = stats + 2048;     // 512
    float* rstd_p  = stats + 2560;     // 512

    // 1) dual GEMM: [q;k] = concat(q_w,k_w) @ x   (bf16 out, fp32 accumulate)
    gemm_kernel<0><<<dim3(N_ / BNT, QKC / BM, TB_), 256, 0, stream>>>(
        q_w, k_w, x, nullptr, (size_t)C_ * N_, qkbuf, nullptr, nullptr);

    // 2) BN stats for q (channels 0-511) and k (512-1023) in one pass
    bn_stats_kernel<__bf16><<<QKC, 256, 0, stream>>>(qkbuf, mean_qk, rstd_qk, QKC);

    // 3) q BN+LIF, head-sum, attn LIF
    q_attn_kernel<<<B_ * H_, 256, 0, stream>>>(qkbuf, mean_qk, rstd_qk,
                                               q_gamma, q_beta, attn);

    // 4) k BN+LIF, y = attn * spikes  (y written into q half of qkbuf)
    k_y_kernel<<<B_ * C_, 256, 0, stream>>>(qkbuf, mean_qk, rstd_qk,
                                            k_gamma, k_beta, attn, qkbuf);

    // 5) proj GEMM: ylin = proj_w @ y + proj_b   (bf16 B, per-batch stride QKC*N)
    gemm_kernel<1><<<dim3(N_ / BNT, C_ / BM, TB_), 256, 0, stream>>>(
        proj_w, nullptr, nullptr, qkbuf, (size_t)QKC * N_, nullptr, ylin, proj_b);

    // 6) BN stats for proj output
    bn_stats_kernel<float><<<C_, 256, 0, stream>>>(ylin, mean_p, rstd_p, C_);

    // 7) final BN + LIF -> f32 spikes
    out_kernel<<<B_ * C_, 256, 0, stream>>>(ylin, mean_p, rstd_p,
                                            proj_gamma, proj_beta, out);
}
